// GRADEBase_88270167867553
// MI455X (gfx1250) — matrix operand order repaired and checked by reading, not compiled, not run on hardware
//
#include <hip/hip_runtime.h>

// ---------------------------------------------------------------------------
// GCN (4x GCNConv + classifier) for MI455X / gfx1250, wave32 + WMMA f16.
// GEMMs use v_wmma_f32_16x16x32_f16 with hi/lo f16 split (near-fp32 accuracy).
// Aggregation uses a dst-CSR gather (no fp atomics on the hot path).
// ---------------------------------------------------------------------------

typedef __attribute__((ext_vector_type(16))) _Float16 v16h;
typedef __attribute__((ext_vector_type(8)))  _Float16 h8;
typedef __attribute__((ext_vector_type(8)))  float    v8f;

#define GCN_N 50000
#define GCN_E 800000
#define GCN_H 512
#define GCN_C 40
#define FEAT_PITCH (4 * GCN_H + GCN_C)   // 2088

#define BM 128
#define BN 64
#define BK 32
#define LDS_S 40                          // padded LDS stride (halfwords)

// ------------------------------- small utils -------------------------------

__global__ void fill_f32_kernel(float* p, float v, int n) {
    int i = blockIdx.x * blockDim.x + threadIdx.x;
    if (i < n) p[i] = v;
}

__global__ void fill_i32_kernel(int* p, int v, int n) {
    int i = blockIdx.x * blockDim.x + threadIdx.x;
    if (i < n) p[i] = v;
}

__global__ void count_deg_kernel(const int* __restrict__ dst, float* __restrict__ deg, int e) {
    int i = blockIdx.x * blockDim.x + threadIdx.x;
    if (i < e) atomicAdd(&deg[dst[i]], 1.0f);
}

__global__ void rsqrt_kernel(const float* __restrict__ deg, float* __restrict__ dinv, int n) {
    int i = blockIdx.x * blockDim.x + threadIdx.x;
    if (i < n) dinv[i] = rsqrtf(deg[i]);   // deg >= 1 always (self-loop)
}

// Exclusive scan of (int)deg -> row_ptr[0..n], single workgroup (n is small).
__global__ void scan_excl_kernel(const float* __restrict__ deg, int* __restrict__ row_ptr, int n) {
    __shared__ int sh[1024];
    __shared__ int carry_s;
    const int tid = threadIdx.x;
    if (tid == 0) carry_s = 0;
    __syncthreads();
    for (int base = 0; base < n; base += 1024) {
        int i = base + tid;
        int v = (i < n) ? (int)deg[i] : 0;
        sh[tid] = v;
        __syncthreads();
        for (int off = 1; off < 1024; off <<= 1) {
            int t = (tid >= off) ? sh[tid - off] : 0;
            __syncthreads();
            sh[tid] += t;
            __syncthreads();
        }
        int incl  = sh[tid];
        int carry = carry_s;
        if (i < n) row_ptr[i] = carry + incl - v;
        __syncthreads();
        if (tid == 1023) carry_s = carry + sh[1023];
        __syncthreads();
    }
    if (tid == 0) row_ptr[n] = carry_s;
}

// Place edges (and self-loops) into dst-CSR with normalization weights.
__global__ void csr_fill_kernel(const int* __restrict__ src, const int* __restrict__ dst,
                                const int* __restrict__ row_ptr, int* __restrict__ cursor,
                                const float* __restrict__ dinv,
                                int* __restrict__ col, float* __restrict__ wgt,
                                int e, int n) {
    int i = blockIdx.x * blockDim.x + threadIdx.x;
    if (i >= e + n) return;
    int s, d;
    if (i < e) { s = src[i]; d = dst[i]; }
    else       { s = i - e;  d = i - e;  }   // self-loop
    int pos = row_ptr[d] + atomicAdd(&cursor[d], 1);
    col[pos] = s;
    wgt[pos] = dinv[s] * dinv[d];
}

// ------------------------------- WMMA GEMM ---------------------------------
// out = A[M x 512] * W[512 x ncols]   (f16 hi/lo split, fp32 accumulate)
// Block: 256 threads = 8 waves; tile BM=128 x BN=64, K-step 32.

__global__ __launch_bounds__(256) void gemm_wmma_kernel(
    const float* __restrict__ A, int lda,
    const float* __restrict__ W, int ncols,
    float* __restrict__ out0, int p0,
    float* __restrict__ out1, int p1,
    const float* __restrict__ bias, int do_relu, int M)
{
    __shared__ __align__(16) _Float16 sAh[BM][LDS_S];
    __shared__ __align__(16) _Float16 sAl[BM][LDS_S];
    __shared__ __align__(16) _Float16 sBh[BN][LDS_S];
    __shared__ __align__(16) _Float16 sBl[BN][LDS_S];

    const int tid     = threadIdx.x;
    const int lane    = tid & 31;
    const int wv      = tid >> 5;
    const int rowBase = blockIdx.x * BM;
    const int colBase = blockIdx.y * BN;

    v8f acc[4];
    #pragma unroll
    for (int ct = 0; ct < 4; ++ct)
        #pragma unroll
        for (int r = 0; r < 8; ++r) acc[ct][r] = 0.0f;

    for (int kk = 0; kk < GCN_H; kk += BK) {
        // ---- stage A tile (128x32 fp32 -> hi/lo f16) ----
        #pragma unroll
        for (int it = 0; it < 4; ++it) {
            int f  = tid + it * 256;          // 0..1023
            int r  = f >> 3;                  // 0..127
            int c4 = (f & 7) * 4;             // 0,4,..,28
            int gr = rowBase + r;
            float4 v;
            if (gr < M) v = *(const float4*)(A + (size_t)gr * lda + kk + c4);
            else        v = make_float4(0.f, 0.f, 0.f, 0.f);
            float xs[4] = {v.x, v.y, v.z, v.w};
            #pragma unroll
            for (int j = 0; j < 4; ++j) {
                _Float16 hi = (_Float16)xs[j];
                sAh[r][c4 + j] = hi;
                sAl[r][c4 + j] = (_Float16)(xs[j] - (float)hi);
            }
        }
        // ---- stage B tile (32x64 fp32, stored K-contiguous / transposed) ----
        #pragma unroll
        for (int it = 0; it < 2; ++it) {
            int f  = tid + it * 256;          // 0..511
            int kr = f >> 4;                  // 0..31
            int c4 = (f & 15) * 4;            // 0,4,..,60
            #pragma unroll
            for (int j = 0; j < 4; ++j) {
                int c = colBase + c4 + j;
                float x = (c < ncols) ? W[(size_t)(kk + kr) * ncols + c] : 0.0f;
                _Float16 hi = (_Float16)x;
                sBh[c4 + j][kr] = hi;
                sBl[c4 + j][kr] = (_Float16)(x - (float)hi);
            }
        }
        __syncthreads();

        // ---- A fragments (ISA 16-bit A 16x32 layout) ----
        const int ar = wv * 16 + (lane & 15);
        const int ka = (lane < 16) ? 0 : 8;   // lanes 0-15: K{0..7,16..23}; 16-31: K{8..15,24..31}
        union { v16h v; h8 h[2]; } ua;
        ua.h[0] = *(const h8*)&sAh[ar][ka];
        ua.h[1] = *(const h8*)&sAh[ar][ka + 16];
        v16h ah = ua.v;
        ua.h[0] = *(const h8*)&sAl[ar][ka];
        ua.h[1] = *(const h8*)&sAl[ar][ka + 16];
        v16h al = ua.v;

        const int kb = (lane < 16) ? 0 : 8;   // B: lanes 0-15 hold K{0..7,16..23}; 16-31: K{8..15,24..31}
        #pragma unroll
        for (int ct = 0; ct < 4; ++ct) {
            int bc = ct * 16 + (lane & 15);
            ua.h[0] = *(const h8*)&sBh[bc][kb];
            ua.h[1] = *(const h8*)&sBh[bc][kb + 16];
            v16h bh = ua.v;
            ua.h[0] = *(const h8*)&sBl[bc][kb];
            ua.h[1] = *(const h8*)&sBl[bc][kb + 16];
            v16h bl = ua.v;
            // x@W ~= hi@Whi + hi@Wlo + lo@Whi  (lo@Wlo ~ 2^-22, dropped)
            acc[ct] = __builtin_amdgcn_wmma_f32_16x16x32_f16(false, ah, false, bh,
                                                             (short)0, acc[ct], false, false);
            acc[ct] = __builtin_amdgcn_wmma_f32_16x16x32_f16(false, ah, false, bl,
                                                             (short)0, acc[ct], false, false);
            acc[ct] = __builtin_amdgcn_wmma_f32_16x16x32_f16(false, al, false, bh,
                                                             (short)0, acc[ct], false, false);
        }
        __syncthreads();
    }

    // ---- epilogue: C layout (VGPR r -> M = r or r+8; N = lane&15) ----
    const int mBase = rowBase + wv * 16 + ((lane < 16) ? 0 : 8);
    const int lcol  = lane & 15;
    #pragma unroll
    for (int ct = 0; ct < 4; ++ct) {
        int c = colBase + ct * 16 + lcol;
        if (c >= ncols) continue;
        float bv = bias ? bias[c] : 0.0f;
        #pragma unroll
        for (int r = 0; r < 8; ++r) {
            int row = mBase + r;
            if (row < M) {
                float val = acc[ct][r] + bv;
                if (do_relu) val = fmaxf(val, 0.0f);
                out0[(size_t)row * p0 + c] = val;
                if (out1) out1[(size_t)row * p1 + c] = val;
            }
        }
    }
}

// ------------------------------ aggregation --------------------------------
// h'[v] = relu( sum_{e in CSR row v} wgt[e] * t[col[e]] + bias );  one block/node.

__global__ __launch_bounds__(128) void aggregate_kernel(
    const float* __restrict__ t, const int* __restrict__ row_ptr,
    const int* __restrict__ col, const float* __restrict__ wgt,
    const float* __restrict__ bias, float* __restrict__ out, int pitch)
{
    const int v = blockIdx.x;
    const int c = threadIdx.x * 4;            // 128 threads x float4 = 512 feats
    float a0 = 0.f, a1 = 0.f, a2 = 0.f, a3 = 0.f;
    const int beg = row_ptr[v], end = row_ptr[v + 1];
    for (int e = beg; e < end; ++e) {
        int   u = col[e];
        float w = wgt[e];
        if (e + 1 < end)
            __builtin_prefetch(t + (size_t)col[e + 1] * GCN_H + c, 0, 1);
        float4 tv = *(const float4*)(t + (size_t)u * GCN_H + c);
        a0 += w * tv.x; a1 += w * tv.y; a2 += w * tv.z; a3 += w * tv.w;
    }
    a0 = fmaxf(a0 + bias[c + 0], 0.f);
    a1 = fmaxf(a1 + bias[c + 1], 0.f);
    a2 = fmaxf(a2 + bias[c + 2], 0.f);
    a3 = fmaxf(a3 + bias[c + 3], 0.f);
    *(float4*)(out + (size_t)v * pitch + c) = make_float4(a0, a1, a2, a3);
}

// ------------------------------- launcher ----------------------------------

extern "C" void kernel_launch(void* const* d_in, const int* in_sizes, int n_in,
                              void* d_out, int out_size, void* d_ws, size_t ws_size,
                              hipStream_t stream) {
    (void)in_sizes; (void)n_in; (void)out_size; (void)ws_size;

    const float* x    = (const float*)d_in[0];           // [N, 512]
    const int*   ei   = (const int*)d_in[1];             // [2, E]
    const float* Wl[4] = {(const float*)d_in[2], (const float*)d_in[4],
                          (const float*)d_in[6], (const float*)d_in[8]};
    const float* bl[4] = {(const float*)d_in[3], (const float*)d_in[5],
                          (const float*)d_in[7], (const float*)d_in[9]};
    const float* clsW = (const float*)d_in[10];          // [512, 40]
    const float* clsb = (const float*)d_in[11];          // [40]

    const int* src = ei;
    const int* dst = ei + GCN_E;

    float* out     = (float*)d_out;                      // [N*C] logits ++ [N*2088] feat_cat
    float* featcat = out + (size_t)GCN_N * GCN_C;

    // ---- workspace carve-out (256B aligned) ----
    char*  ws  = (char*)d_ws;
    size_t off = 0;
    auto carve = [&](size_t bytes) { void* p = ws + off; off += (bytes + 255) & ~(size_t)255; return p; };
    float* t       = (float*)carve((size_t)GCN_N * GCN_H * sizeof(float)); // 102.4 MB
    float* deg     = (float*)carve((size_t)GCN_N * sizeof(float));
    float* dinv    = (float*)carve((size_t)GCN_N * sizeof(float));
    int*   row_ptr = (int*)  carve((size_t)(GCN_N + 1) * sizeof(int));
    int*   cursor  = (int*)  carve((size_t)GCN_N * sizeof(int));
    int*   csr_col = (int*)  carve((size_t)(GCN_E + GCN_N) * sizeof(int));
    float* csr_wgt = (float*)carve((size_t)(GCN_E + GCN_N) * sizeof(float));

    const int EN = GCN_E + GCN_N;

    // ---- graph preprocessing (once per call, reused by all 4 layers) ----
    fill_f32_kernel<<<(GCN_N + 255) / 256, 256, 0, stream>>>(deg, 1.0f, GCN_N);   // self-loop
    fill_i32_kernel<<<(GCN_N + 255) / 256, 256, 0, stream>>>(cursor, 0, GCN_N);
    count_deg_kernel<<<(GCN_E + 255) / 256, 256, 0, stream>>>(dst, deg, GCN_E);
    rsqrt_kernel<<<(GCN_N + 255) / 256, 256, 0, stream>>>(deg, dinv, GCN_N);
    scan_excl_kernel<<<1, 1024, 0, stream>>>(deg, row_ptr, GCN_N);
    csr_fill_kernel<<<(EN + 255) / 256, 256, 0, stream>>>(src, dst, row_ptr, cursor,
                                                          dinv, csr_col, csr_wgt, GCN_E, GCN_N);

    const int mblocks = (GCN_N + BM - 1) / BM;           // 391
    dim3 gemm_grid(mblocks, GCN_H / BN);                 // 391 x 8
    dim3 cls_grid(mblocks, 1);

    // ---- layer 0: t = x @ W0 ; h1 = relu(Agg(t) + b0) -> featcat[:, 0:512] ----
    gemm_wmma_kernel<<<gemm_grid, 256, 0, stream>>>(x, GCN_H, Wl[0], GCN_H,
                                                    t, GCN_H, nullptr, 0, nullptr, 0, GCN_N);
    aggregate_kernel<<<GCN_N, 128, 0, stream>>>(t, row_ptr, csr_col, csr_wgt,
                                                bl[0], featcat, FEAT_PITCH);

    // ---- layers 1..3: read previous h out of feat_cat (pitch 2088) ----
    for (int i = 1; i < 4; ++i) {
        const float* hprev = featcat + (size_t)(i - 1) * GCN_H;
        gemm_wmma_kernel<<<gemm_grid, 256, 0, stream>>>(hprev, FEAT_PITCH, Wl[i], GCN_H,
                                                        t, GCN_H, nullptr, 0, nullptr, 0, GCN_N);
        aggregate_kernel<<<GCN_N, 128, 0, stream>>>(t, row_ptr, csr_col, csr_wgt,
                                                    bl[i], featcat + (size_t)i * GCN_H, FEAT_PITCH);
    }

    // ---- classifier: logits = h4 @ clsW + clsb -> d_out[0:N*40] and feat_cat[:, 2048:2088] ----
    gemm_wmma_kernel<<<cls_grid, 256, 0, stream>>>(featcat + (size_t)3 * GCN_H, FEAT_PITCH,
                                                   clsW, GCN_C,
                                                   out, GCN_C,
                                                   featcat + (size_t)4 * GCN_H, FEAT_PITCH,
                                                   clsb, 0, GCN_N);
}
